// Dueling_QNet_12867722018988
// MI455X (gfx1250) — matrix operand order repaired and checked by reading, not compiled, not run on hardware
//
#include <hip/hip_runtime.h>
#include <hip/hip_bf16.h>

typedef __attribute__((ext_vector_type(16))) _Float16 v16h;
typedef __attribute__((ext_vector_type(8)))  float    v8f;

#define HIDDEN    128
#define K1        32      // 2*PERIODS padded to 32 for one WMMA k-step
#define ROWS      64      // edges per tile
#define TILES     8       // tiles per block -> 512 edges/block, weights staged once
#define THREADS   128     // 4 waves (wave32)
#define A_STRIDE  40      // halves per row of sA (pad 32->40: kills bank conflicts, keeps 16B align)
#define H1_STRIDE 136     // halves per row of sH1 (pad 128->136)

// -------- workspace layout (bytes) --------
// [0]                       float advSum
// [64]                      f16 W1T [128][32]   (W1T[n][k] = W1[k][n], k>=24 -> 0)
// [64+8192]                 f16 W2T [128][128]  (W2T[n][k] = W2[k][n])
// [64+8192+32768]           f16 Wv [128]
// [64+8192+32768+256]       f16 Wa [128]

__global__ void prep_weights(const float* __restrict__ W1,
                             const float* __restrict__ W2,
                             const float* __restrict__ Wv,
                             const float* __restrict__ Wa,
                             _Float16* __restrict__ w1t,
                             _Float16* __restrict__ w2t,
                             _Float16* __restrict__ wvh,
                             _Float16* __restrict__ wah,
                             float* __restrict__ advSum) {
    int t = threadIdx.x + blockIdx.x * blockDim.x;
    int stride = blockDim.x * gridDim.x;
    if (t == 0) *advSum = 0.0f;
    for (int i = t; i < 128 * 32; i += stride) {        // W1T: n-major, K padded
        int n = i >> 5, k = i & 31;
        float v = (k < 24) ? W1[k * HIDDEN + n] : 0.0f;
        w1t[n * 32 + k] = (_Float16)v;
    }
    for (int i = t; i < 128 * 128; i += stride) {       // W2T: n-major
        int n = i >> 7, k = i & 127;
        w2t[n * HIDDEN + k] = (_Float16)W2[k * HIDDEN + n];
    }
    for (int i = t; i < 128; i += stride) {
        wvh[i] = (_Float16)Wv[i];
        wah[i] = (_Float16)Wa[i];
    }
}

__global__ __launch_bounds__(THREADS)
void edge_mlp_wmma(const float* __restrict__ nodeEmb,
                   const int*   __restrict__ srcIdx,
                   const int*   __restrict__ dstIdx,
                   const float* __restrict__ gb1,
                   const float* __restrict__ gb2,
                   const float* __restrict__ gbv,
                   const float* __restrict__ gba,
                   const _Float16* __restrict__ w1t,
                   const _Float16* __restrict__ w2t,
                   const _Float16* __restrict__ wvh,
                   const _Float16* __restrict__ wah,
                   float* __restrict__ out,
                   float* __restrict__ advSum,
                   int E) {
    __shared__ __align__(16) _Float16 sW1[128 * K1];        // 8192 B
    __shared__ __align__(16) _Float16 sW2[128 * HIDDEN];    // 32768 B
    __shared__ __align__(16) _Float16 sWv[128];             // 256 B
    __shared__ __align__(16) _Float16 sWa[128];             // 256 B
    __shared__ __align__(16) _Float16 sA[ROWS * A_STRIDE];  // 5120 B
    __shared__ __align__(16) _Float16 sH1[ROWS * H1_STRIDE];// 17408 B
    __shared__ int   sSrc[ROWS];
    __shared__ int   sDst[ROWS];
    __shared__ float sAdv;

    const int tid    = threadIdx.x;
    const int lane   = tid & 31;
    const int wave   = tid >> 5;
    const int nLane  = lane & 15;       // column within a 16-wide N tile
    const int hiHalf = lane >> 4;       // 0: lanes 0-15, 1: lanes 16-31
    const int blockBase = blockIdx.x * (ROWS * TILES);

    if (tid == 0) sAdv = 0.0f;

    // ---- stage weights into LDS with CDNA5 async global->LDS copies ----
    // GLOBAL_LOAD_ASYNC_TO_LDS_B128 (saddr form): per-lane 16B, no VGPR data,
    // tracked by ASYNCcnt. vdst = LDS byte address, vaddr = 32-bit offset.
    {
        unsigned ldsW1 = (unsigned)(uintptr_t)&sW1[0];
        unsigned ldsW2 = (unsigned)(uintptr_t)&sW2[0];
        unsigned long long gW1 = (unsigned long long)(uintptr_t)w1t;
        unsigned long long gW2 = (unsigned long long)(uintptr_t)w2t;
        for (int i = tid; i < 512; i += THREADS) {          // 8 KB
            unsigned l = ldsW1 + (unsigned)(i * 16);
            unsigned o = (unsigned)(i * 16);
            asm volatile("global_load_async_to_lds_b128 %0, %1, %2"
                         :: "v"(l), "v"(o), "s"(gW1) : "memory");
        }
        for (int i = tid; i < 2048; i += THREADS) {         // 32 KB
            unsigned l = ldsW2 + (unsigned)(i * 16);
            unsigned o = (unsigned)(i * 16);
            asm volatile("global_load_async_to_lds_b128 %0, %1, %2"
                         :: "v"(l), "v"(o), "s"(gW2) : "memory");
        }
    }
    if (tid < 128) {
        sWv[tid] = wvh[tid];
        sWa[tid] = wah[tid];
    }
    asm volatile("s_wait_asynccnt 0" ::: "memory");

    const int mrow  = (wave << 4) + nLane;          // A-fragment row for this lane
    const int mBase = (wave << 4) + (hiHalf << 3);  // first of this lane's 8 C/D rows

    // B-fragment loaders (B layout: lane holds K of column n;
    // lanes 0-15 K{0..7,16..23}, lanes 16-31 K{8..15,24..31} of the k-step)
    auto loadB1 = [&](v16h& dst, int nt) {
        int n = (nt << 4) + nLane;
        const _Float16* p = &sW1[n * K1 + (hiHalf ? 8 : 0)];
        ((uint4*)&dst)[0] = ((const uint4*)p)[0];
        ((uint4*)&dst)[1] = ((const uint4*)p)[2];
    };
    auto loadB2 = [&](v16h& dst, int nt, int kb) {
        int n = (nt << 4) + nLane;
        const _Float16* p = &sW2[n * HIDDEN + kb + (hiHalf ? 8 : 0)];
        ((uint4*)&dst)[0] = ((const uint4*)p)[0];
        ((uint4*)&dst)[1] = ((const uint4*)p)[2];
    };

    for (int tile = 0; tile < TILES; ++tile) {
        const int e0 = blockBase + tile * ROWS;
        if (e0 >= E) break;

        __syncthreads();   // previous tile's sA/sSrc reads complete; weights ready on tile 0

        // ---- stage edge indices (prefetch next tile's index stream) ----
        if (tid < ROWS) {
            int e = e0 + tid;
            sSrc[tid] = (e < E) ? srcIdx[e] : 0;
            if (e0 + ROWS < E) __builtin_prefetch(&srcIdx[e0 + ROWS + tid], 0, 0);
        } else if (tid < 2 * ROWS) {
            int e = e0 + tid - ROWS;
            sDst[tid - ROWS] = (e < E) ? dstIdx[e] : 0;
            if (e0 + ROWS < E) __builtin_prefetch(&dstIdx[e0 + tid], 0, 0);
        }
        __syncthreads();

        // ---- gather node embeddings -> f16 A tile [64][32] (cols 0-11 src, 12-23 dst, 24-31 zero)
        for (int i = tid; i < ROWS * K1; i += THREADS) {
            int r = i >> 5, c = i & 31;
            float v = 0.0f;
            if (c < 12)      v = nodeEmb[sSrc[r] * 12 + c];
            else if (c < 24) v = nodeEmb[sDst[r] * 12 + (c - 12)];
            sA[r * A_STRIDE + c] = (_Float16)v;
        }
        __syncthreads();

        // ================= GEMM1: [64,32] x [32,128], single k-step =================
        v16h aFrag;
        {
            // A 16-bit 16x32 layout: lanes 0-15 hold K{0..7,16..23}, lanes 16-31 K{8..15,24..31}
            const _Float16* p = &sA[mrow * A_STRIDE + (hiHalf ? 8 : 0)];
            ((uint4*)&aFrag)[0] = *(const uint4*)(p);
            ((uint4*)&aFrag)[1] = *(const uint4*)(p + 16);
        }
        {
            v16h bBuf0, bBuf1;                 // double buffer: next load issued before wmma
            loadB1(bBuf0, 0);
            #pragma unroll
            for (int nt = 0; nt < 8; ++nt) {
                v16h& cur = (nt & 1) ? bBuf1 : bBuf0;
                v16h& nxt = (nt & 1) ? bBuf0 : bBuf1;
                if (nt < 7) loadB1(nxt, nt + 1);
                v8f c = {};
                c = __builtin_amdgcn_wmma_f32_16x16x32_f16(false, aFrag, false, cur,
                                                           (short)0, c, false, false);
                int n = (nt << 4) + nLane;
                float bias = gb1[n];
                #pragma unroll
                for (int r = 0; r < 8; ++r) {
                    float h = c[r] + bias;
                    h = h > 0.0f ? h : 0.0f;
                    sH1[(mBase + r) * H1_STRIDE + n] = (_Float16)h;
                }
            }
        }
        // No barrier: each wave wrote rows [wave*16, wave*16+16) and reads only those rows.

        // ================= GEMM2: [64,128] x [128,128], 4 k-steps, 8 live acc tiles ============
        v8f acc[8];
        v8f zero = {};
        #pragma unroll
        for (int nt = 0; nt < 8; ++nt) acc[nt] = zero;

        for (int kt = 0; kt < 4; ++kt) {
            int kb = kt << 5;
            v16h a2;
            const _Float16* pa = &sH1[mrow * H1_STRIDE + kb + (hiHalf ? 8 : 0)];
            ((uint4*)&a2)[0] = *(const uint4*)(pa);
            ((uint4*)&a2)[1] = *(const uint4*)(pa + 16);

            v16h bBuf0, bBuf1;                 // double buffer: next load issued before wmma
            loadB2(bBuf0, 0, kb);
            #pragma unroll
            for (int nt = 0; nt < 8; ++nt) {
                v16h& cur = (nt & 1) ? bBuf1 : bBuf0;
                v16h& nxt = (nt & 1) ? bBuf0 : bBuf1;
                if (nt < 7) loadB2(nxt, nt + 1, kb);
                acc[nt] = __builtin_amdgcn_wmma_f32_16x16x32_f16(false, a2, false, cur,
                                                                 (short)0, acc[nt], false, false);
            }
        }

        // ================= dueling heads, fused in registers =================
        float vAcc[8] = {0,0,0,0,0,0,0,0};
        float aAcc[8] = {0,0,0,0,0,0,0,0};
        #pragma unroll
        for (int nt = 0; nt < 8; ++nt) {
            int n = (nt << 4) + nLane;
            float bias = gb2[n];
            float wv = (float)sWv[n];
            float wa = (float)sWa[n];
            #pragma unroll
            for (int r = 0; r < 8; ++r) {
                float h = acc[nt][r] + bias;
                h = h > 0.0f ? h : 0.0f;
                vAcc[r] += h * wv;
                aAcc[r] += h * wa;
            }
        }
        // reduce over the 16 lanes of each half-wave (rows differ between halves)
        #pragma unroll
        for (int off = 1; off < 16; off <<= 1) {
            #pragma unroll
            for (int r = 0; r < 8; ++r) {
                vAcc[r] += __shfl_xor(vAcc[r], off, 32);
                aAcc[r] += __shfl_xor(aAcc[r], off, 32);
            }
        }
        if (nLane == 0) {   // lanes 0 and 16 hold full sums for their 8 rows
            float bv = gbv[0], ba = gba[0];
            float advLocal = 0.0f;
            #pragma unroll
            for (int r = 0; r < 8; ++r) {
                int e = e0 + mBase + r;
                if (e < E) {
                    float val = vAcc[r] + bv;
                    float adv = aAcc[r] + ba;
                    out[e] = val + adv;       // mean(adv) subtracted in pass 2
                    advLocal += adv;
                }
            }
            atomicAdd(&sAdv, advLocal);
        }
    }

    __syncthreads();
    if (tid == 0) atomicAdd(advSum, sAdv);
}

__global__ void sub_mean(float* __restrict__ out,
                         const float* __restrict__ advSum, int E) {
    int i = blockIdx.x * blockDim.x + threadIdx.x;
    float m = advSum[0] / (float)E;
    if (i < E) out[i] -= m;
}

extern "C" void kernel_launch(void* const* d_in, const int* in_sizes, int n_in,
                              void* d_out, int out_size, void* d_ws, size_t ws_size,
                              hipStream_t stream) {
    const float* nodeEmb = (const float*)d_in[0];
    const int*   src     = (const int*)d_in[1];
    const int*   dst     = (const int*)d_in[2];
    const float* W1      = (const float*)d_in[3];
    const float* b1      = (const float*)d_in[4];
    const float* W2      = (const float*)d_in[5];
    const float* b2      = (const float*)d_in[6];
    const float* Wv      = (const float*)d_in[7];
    const float* bv      = (const float*)d_in[8];
    const float* Wa      = (const float*)d_in[9];
    const float* ba      = (const float*)d_in[10];
    const int E = in_sizes[1];

    char* ws = (char*)d_ws;
    float*    advSum = (float*)ws;
    _Float16* w1t = (_Float16*)(ws + 64);
    _Float16* w2t = (_Float16*)(ws + 64 + 8192);
    _Float16* wvh = (_Float16*)(ws + 64 + 8192 + 32768);
    _Float16* wah = (_Float16*)(ws + 64 + 8192 + 32768 + 256);

    prep_weights<<<32, 256, 0, stream>>>(W1, W2, Wv, Wa, w1t, w2t, wvh, wah, advSum);

    int edgesPerBlock = ROWS * TILES;
    int nBlocks = (E + edgesPerBlock - 1) / edgesPerBlock;
    edge_mlp_wmma<<<nBlocks, THREADS, 0, stream>>>(nodeEmb, src, dst, b1, b2, bv, ba,
                                                   w1t, w2t, wvh, wah,
                                                   (float*)d_out, advSum, E);

    sub_mean<<<(E + 255) / 256, 256, 0, stream>>>((float*)d_out, advSum, E);
}
